// GAT19Model_6124623364727
// MI455X (gfx1250) — matrix operand order repaired and checked by reading, not compiled, not run on hardware
//
#include <hip/hip_runtime.h>
#include <cstdint>

#define WG 256
#define SLOPE 0.2f
#define BN_EPS 1e-5f
#define WMAX 4096  // max padded weight elements per matrix: layer1 = 32*128

typedef __attribute__((ext_vector_type(16))) _Float16 v16h;
typedef __attribute__((ext_vector_type(8)))  float    v8f;

__device__ __forceinline__ unsigned f2ord(float f) {
  unsigned u = __float_as_uint(f);
  return (u & 0x80000000u) ? ~u : (u | 0x80000000u);
}
__device__ __forceinline__ float ord2f(unsigned o) {
  unsigned u = (o & 0x80000000u) ? (o ^ 0x80000000u) : ~o;
  return __uint_as_float(u);
}

__global__ void fill_u32_kernel(unsigned* __restrict__ p, unsigned v, long n) {
  long i = (long)blockIdx.x * blockDim.x + threadIdx.x;
  if (i < n) p[i] = v;
}

// xl = x @ Wl^T, xr = x @ Wr^T ; one wave per 16-row tile of x.
// Weights are staged once per block into LDS as fp16, zero-padded to otiles*16
// columns, so the WMMA loop has no divergent guards: A comes in as four
// global b128 loads per k-step, B as two LDS b128 loads per fragment.
__global__ void node_transform_wmma(const float* __restrict__ x,
                                    const float* __restrict__ Wl,
                                    const float* __restrict__ Wr,
                                    float* __restrict__ xl,
                                    float* __restrict__ xr,
                                    int ntiles, int K, int Cout) {
  __shared__ alignas(32) _Float16 wl_s[WMAX];
  __shared__ alignas(32) _Float16 wr_s[WMAX];

  const int otiles = (Cout + 15) >> 4;
  const int cpad   = otiles << 4;           // Cout padded to multiple of 16
  const int wtot   = cpad * K;

  // cooperative stage: fp32 -> fp16, zero rows for o >= Cout
  for (int idx = threadIdx.x; idx < wtot; idx += blockDim.x) {
    const int o = idx / K, k = idx - o * K;
    float vl = 0.f, vr = 0.f;
    if (o < Cout) {
      vl = Wl[(size_t)o * K + k];
      vr = Wr[(size_t)o * K + k];
    }
    wl_s[idx] = (_Float16)vl;
    wr_s[idx] = (_Float16)vr;
  }
  __syncthreads();

  const int tile = blockIdx.x * (blockDim.x >> 5) + (threadIdx.x >> 5);
  if (tile >= ntiles) return;               // wave-uniform: EXEC all-ones below
  const int lane   = threadIdx.x & 31;
  const int lane16 = lane & 15;
  const int hi     = lane >> 4;             // lane half selects K sub-ranges
  const float* __restrict__ xrow = x + (size_t)((tile << 4) + lane16) * K;

  for (int ot = 0; ot < otiles; ++ot) {
    const int ocol = (ot << 4) + lane16;    // B column held by this lane
    const _Float16* __restrict__ blrow = wl_s + (size_t)ocol * K;
    const _Float16* __restrict__ brrow = wr_s + (size_t)ocol * K;
    v8f accL = {};
    v8f accR = {};
    for (int kb = 0; kb < K; kb += 32) {
      // A 16x32 fp16 layout: halves 0-15 -> k = kb+16*hi+0..15 (contiguous)
      const int k0 = kb + (hi << 4);
      const float4 f0 = *(const float4*)(xrow + k0);
      const float4 f1 = *(const float4*)(xrow + k0 + 4);
      const float4 f2 = *(const float4*)(xrow + k0 + 8);
      const float4 f3 = *(const float4*)(xrow + k0 + 12);
      v16h a;
      a[0]  = (_Float16)f0.x; a[1]  = (_Float16)f0.y;
      a[2]  = (_Float16)f0.z; a[3]  = (_Float16)f0.w;
      a[4]  = (_Float16)f1.x; a[5]  = (_Float16)f1.y;
      a[6]  = (_Float16)f1.z; a[7]  = (_Float16)f1.w;
      a[8]  = (_Float16)f2.x; a[9]  = (_Float16)f2.y;
      a[10] = (_Float16)f2.z; a[11] = (_Float16)f2.w;
      a[12] = (_Float16)f3.x; a[13] = (_Float16)f3.y;
      a[14] = (_Float16)f3.z; a[15] = (_Float16)f3.w;
      // B 32x16 fp16 layout: halves 0-15 -> k = kb+16*hi+0..15 (contiguous)
      const v16h bl = *(const v16h*)(blrow + kb + (hi << 4));
      const v16h br = *(const v16h*)(brrow + kb + (hi << 4));
      accL = __builtin_amdgcn_wmma_f32_16x16x32_f16(false, a, false, bl,
                                                    (short)0, accL, false, false);
      accR = __builtin_amdgcn_wmma_f32_16x16x32_f16(false, a, false, br,
                                                    (short)0, accR, false, false);
    }
    const int row0 = (tile << 4) + (hi << 3); // D row = r + 8*hi
    if (ocol < Cout) {                        // divergent only at store
#pragma unroll
      for (int r = 0; r < 8; ++r) {
        xl[(size_t)(row0 + r) * Cout + ocol] = accL[r];
        xr[(size_t)(row0 + r) * Cout + ocol] = accR[r];
      }
    }
  }
}

// Pass 1: s_e = att . leaky_relu(xl[src] + xr[dst] + ea*We), atomicMax into m[dst]
__global__ void edge_score_kernel(const float* __restrict__ xl,
                                  const float* __restrict__ xr,
                                  const float* __restrict__ ea,
                                  const int* __restrict__ src,
                                  const int* __restrict__ dst,
                                  const float* __restrict__ We,
                                  const float* __restrict__ att,
                                  float* __restrict__ s,
                                  unsigned* __restrict__ mo,
                                  int E, int C) {
  int e = blockIdx.x * blockDim.x + threadIdx.x;
  if (e >= E) return;
  const int si = src[e], di = dst[e];
  const float w = ea[e];
  const float* pl = xl + (size_t)si * C;
  const float* pr = xr + (size_t)di * C;
  float acc = 0.f;
  if ((C & 3) == 0) {
    const float4* pl4 = (const float4*)pl;
    const float4* pr4 = (const float4*)pr;
    const float4* we4 = (const float4*)We;
    const float4* at4 = (const float4*)att;
    const int n4 = C >> 2;
    for (int c = 0; c < n4; ++c) {
      float4 l = pl4[c], r = pr4[c], wv = we4[c], av = at4[c];
      float v0 = l.x + r.x + w * wv.x; v0 = fmaxf(v0, SLOPE * v0);
      float v1 = l.y + r.y + w * wv.y; v1 = fmaxf(v1, SLOPE * v1);
      float v2 = l.z + r.z + w * wv.z; v2 = fmaxf(v2, SLOPE * v2);
      float v3 = l.w + r.w + w * wv.w; v3 = fmaxf(v3, SLOPE * v3);
      acc += av.x * v0 + av.y * v1 + av.z * v2 + av.w * v3;
    }
  } else {
    for (int c = 0; c < C; ++c) {
      float v = pl[c] + pr[c] + w * We[c];
      v = fmaxf(v, SLOPE * v);
      acc += att[c] * v;
    }
  }
  s[e] = acc;
  atomicMax(&mo[di], f2ord(acc));
}

// Pass 2: a_e = exp(s_e - m[dst]); d[dst] += a_e; s <- a
__global__ void edge_exp_kernel(float* __restrict__ s,
                                const unsigned* __restrict__ mo,
                                const int* __restrict__ dst,
                                float* __restrict__ d, int E) {
  int e = blockIdx.x * blockDim.x + threadIdx.x;
  if (e >= E) return;
  const int di = dst[e];
  const float a = expf(s[e] - ord2f(mo[di]));
  s[e] = a;
  atomicAdd(&d[di], a);
}

__global__ void node_invd_kernel(float* __restrict__ d, int N) {
  int i = blockIdx.x * blockDim.x + threadIdx.x;
  if (i < N) {
    float v = d[i];
    d[i] = 1.0f / (v == 0.0f ? 1.0f : v);
  }
}

// Pass 3: acc[dst] += (a_e / d[dst]) * xl[src]
__global__ void edge_aggregate_kernel(const float* __restrict__ s,
                                      const float* __restrict__ invd,
                                      const float* __restrict__ xl,
                                      const int* __restrict__ src,
                                      const int* __restrict__ dst,
                                      float* __restrict__ acc,
                                      int E, int C) {
  int e = blockIdx.x * blockDim.x + threadIdx.x;
  if (e >= E) return;
  const int si = src[e], di = dst[e];
  const float w = s[e] * invd[di];
  const float* pl = xl + (size_t)si * C;
  float* pa = acc + (size_t)di * C;
  if ((C & 3) == 0) {
    const float4* pl4 = (const float4*)pl;
    const int n4 = C >> 2;
    for (int c = 0; c < n4; ++c) {
      float4 l = pl4[c];
      atomicAdd(&pa[4 * c + 0], w * l.x);
      atomicAdd(&pa[4 * c + 1], w * l.y);
      atomicAdd(&pa[4 * c + 2], w * l.z);
      atomicAdd(&pa[4 * c + 3], w * l.w);
    }
  } else {
    for (int c = 0; c < C; ++c) atomicAdd(&pa[c], w * pl[c]);
  }
}

__global__ void finalize_bias_kernel(const float* __restrict__ acc,
                                     const float* __restrict__ bias,
                                     float* __restrict__ out, long n, int C) {
  long i = (long)blockIdx.x * blockDim.x + threadIdx.x;
  if (i < n) out[i] = acc[i] + bias[i % C];
}

// One block per channel: biased mean/var -> scale/shift
__global__ void bn_stats_kernel(const float* __restrict__ x,
                                const float* __restrict__ g,
                                const float* __restrict__ b,
                                float* __restrict__ ss, int N, int C) {
  const int c = blockIdx.x;
  __shared__ float s0[WG], s1[WG];
  float a0 = 0.f, a1 = 0.f;
  for (int n = threadIdx.x; n < N; n += blockDim.x) {
    float v = x[(size_t)n * C + c];
    a0 += v; a1 += v * v;
  }
  s0[threadIdx.x] = a0; s1[threadIdx.x] = a1;
  __syncthreads();
  for (int st = WG / 2; st > 0; st >>= 1) {
    if (threadIdx.x < st) {
      s0[threadIdx.x] += s0[threadIdx.x + st];
      s1[threadIdx.x] += s1[threadIdx.x + st];
    }
    __syncthreads();
  }
  if (threadIdx.x == 0) {
    float mu  = s0[0] / (float)N;
    float var = s1[0] / (float)N - mu * mu;
    float sc  = g[c] * rsqrtf(var + BN_EPS);
    ss[c]     = sc;
    ss[C + c] = b[c] - mu * sc;
  }
}

__global__ void bn_apply_kernel(float* __restrict__ x, const float* __restrict__ ss,
                                long n, int C) {
  long i = (long)blockIdx.x * blockDim.x + threadIdx.x;
  if (i < n) {
    int c = (int)(i % C);
    x[i] = x[i] * ss[c] + ss[C + c];
  }
}

__global__ void mean_reduce_kernel(const float* __restrict__ x, float* __restrict__ out,
                                   int N) {
  __shared__ float sm[WG];
  float s = 0.f;
  for (int i = blockIdx.x * blockDim.x + threadIdx.x; i < N;
       i += gridDim.x * blockDim.x)
    s += x[i];
  sm[threadIdx.x] = s;
  __syncthreads();
  for (int st = WG / 2; st > 0; st >>= 1) {
    if (threadIdx.x < st) sm[threadIdx.x] += sm[threadIdx.x + st];
    __syncthreads();
  }
  if (threadIdx.x == 0) atomicAdd(out, sm[0] / (float)N);
}

static inline int cdiv(long a, int b) { return (int)((a + (long)b - 1) / b); }

static void run_gat_layer(const float* x_in, int Cin, int Cout,
                          const float* Wl, const float* Wr, const float* We,
                          const float* att, const float* bias,
                          const float* g, const float* b,
                          float* x_out,
                          float* xl, float* xr_acc, float* sb,
                          unsigned* mo, float* db, float* ss,
                          const int* src, const int* dst, const float* ea,
                          int N, int E, hipStream_t stream) {
  const int ntiles = N >> 4;  // N divisible by 16 (100000 = 6250*16)
  node_transform_wmma<<<cdiv(ntiles, WG / 32), WG, 0, stream>>>(
      x_in, Wl, Wr, xl, xr_acc, ntiles, Cin, Cout);
  fill_u32_kernel<<<cdiv(N, WG), WG, 0, stream>>>(mo, 0x007FFFFFu, N); // ord(-inf)
  fill_u32_kernel<<<cdiv(N, WG), WG, 0, stream>>>((unsigned*)db, 0u, N);
  edge_score_kernel<<<cdiv(E, WG), WG, 0, stream>>>(xl, xr_acc, ea, src, dst, We,
                                                    att, sb, mo, E, Cout);
  edge_exp_kernel<<<cdiv(E, WG), WG, 0, stream>>>(sb, mo, dst, db, E);
  // xr is dead after edge_score -> reuse its buffer as the output accumulator
  fill_u32_kernel<<<cdiv((long)N * Cout, WG), WG, 0, stream>>>(
      (unsigned*)xr_acc, 0u, (long)N * Cout);
  node_invd_kernel<<<cdiv(N, WG), WG, 0, stream>>>(db, N);
  edge_aggregate_kernel<<<cdiv(E, WG), WG, 0, stream>>>(sb, db, xl, src, dst,
                                                        xr_acc, E, Cout);
  finalize_bias_kernel<<<cdiv((long)N * Cout, WG), WG, 0, stream>>>(
      xr_acc, bias, x_out, (long)N * Cout, Cout);
  if (g) {
    bn_stats_kernel<<<Cout, WG, 0, stream>>>(x_out, g, b, ss, N, Cout);
    bn_apply_kernel<<<cdiv((long)N * Cout, WG), WG, 0, stream>>>(
        x_out, ss, (long)N * Cout, Cout);
  }
}

extern "C" void kernel_launch(void* const* d_in, const int* in_sizes, int n_in,
                              void* d_out, int out_size, void* d_ws, size_t ws_size,
                              hipStream_t stream) {
  (void)n_in; (void)out_size; (void)ws_size;
  const float* features  = (const float*)d_in[0];
  const float* edge_attr = (const float*)d_in[1];
  const int*   edge_idx  = (const int*)d_in[2];
  const int N = in_sizes[0] / 128;
  const int E = in_sizes[1];
  const int* src = edge_idx;
  const int* dst = edge_idx + E;

  const float* Wl[4]  = {(const float*)d_in[3],  (const float*)d_in[8],
                         (const float*)d_in[13], (const float*)d_in[18]};
  const float* Wr[4]  = {(const float*)d_in[4],  (const float*)d_in[9],
                         (const float*)d_in[14], (const float*)d_in[19]};
  const float* We[4]  = {(const float*)d_in[5],  (const float*)d_in[10],
                         (const float*)d_in[15], (const float*)d_in[20]};
  const float* att[4] = {(const float*)d_in[6],  (const float*)d_in[11],
                         (const float*)d_in[16], (const float*)d_in[21]};
  const float* bia[4] = {(const float*)d_in[7],  (const float*)d_in[12],
                         (const float*)d_in[17], (const float*)d_in[22]};
  const float* bng[3] = {(const float*)d_in[23], (const float*)d_in[25],
                         (const float*)d_in[27]};
  const float* bnb[3] = {(const float*)d_in[24], (const float*)d_in[26],
                         (const float*)d_in[28]};
  const int Cin[4]  = {128, 32, 64, 32};
  const int Cout[4] = {32, 64, 32, 1};

  // Workspace carve (floats): xl | xr/acc | xA | xB | s | m | d | scale/shift
  const size_t N64 = (size_t)N * 64;
  float*    xl = (float*)d_ws;
  float*    xr = xl + N64;
  float*    xA = xr + N64;
  float*    xB = xA + N64;
  float*    sb = xB + N64;            // E floats (edge scores -> softmax weights)
  unsigned* mo = (unsigned*)(sb + E); // N ordered-float maxima
  float*    db = (float*)(mo + N);    // N softmax denominators -> reciprocals
  float*    ss = db + N;              // 2*C scale/shift

  // Layer 1: features -> xA ; Layer 2: xA -> xB ; Layer 3: xB -> xA ;
  // Layer 4: xA -> sb (N floats, C=1)
  run_gat_layer(features, Cin[0], Cout[0], Wl[0], Wr[0], We[0], att[0], bia[0],
                bng[0], bnb[0], xA, xl, xr, sb, mo, db, ss, src, dst, edge_attr,
                N, E, stream);
  run_gat_layer(xA, Cin[1], Cout[1], Wl[1], Wr[1], We[1], att[1], bia[1],
                bng[1], bnb[1], xB, xl, xr, sb, mo, db, ss, src, dst, edge_attr,
                N, E, stream);
  run_gat_layer(xB, Cin[2], Cout[2], Wl[2], Wr[2], We[2], att[2], bia[2],
                bng[2], bnb[2], xA, xl, xr, sb, mo, db, ss, src, dst, edge_attr,
                N, E, stream);
  run_gat_layer(xA, Cin[3], Cout[3], Wl[3], Wr[3], We[3], att[3], bia[3],
                nullptr, nullptr, sb, xl, xr, sb, mo, db, ss, src, dst, edge_attr,
                N, E, stream);

  fill_u32_kernel<<<1, 1, 0, stream>>>((unsigned*)d_out, 0u, 1);
  mean_reduce_kernel<<<240, WG, 0, stream>>>(sb, (float*)d_out, N);
}
